// VQuantizer_13924283973822
// MI455X (gfx1250) — matrix operand order repaired and checked by reading, not compiled, not run on hardware
//
#include <hip/hip_runtime.h>
#include <hip/hip_bf16.h>

// ---------------------------------------------------------------------------
// VQ-VAE vector quantizer for MI455X (gfx1250, wave32, WMMA).
// Heavy path: 16384x8192x256 GEMM in bf16 WMMA, fused with per-row argmin of
// e2[k] - 2*(z.e). Codebook tiles are staged into LDS with CDNA5 async
// global->LDS copies (ASYNCcnt), triple-buffered 2 iterations ahead so L2
// latency hides behind the matrix pipe. Final gather/loss in fp32.
// ---------------------------------------------------------------------------

typedef __attribute__((ext_vector_type(16))) __bf16 v16bf;
typedef __attribute__((ext_vector_type(8)))  float  v8f;

#define K_ENTRIES 8192
#define CDIM      256
#define NVEC      16384       // 16*32*32
#define ZTOTAL    4194304     // 16*32*32*256

#define ROW_US    264         // LDS row stride in ushorts (512B + 16B pad)
#define BUF_US    (16 * ROW_US)
#define BUF_BYTES (BUF_US * 2)   // 8448 B per buffer
#define NIT       (K_ENTRIES / 16)

__device__ __forceinline__ unsigned short f32_to_bf16_rne(float x) {
    unsigned u = __float_as_uint(x);
    unsigned r = u + 0x7FFFu + ((u >> 16) & 1u);   // round-to-nearest-even
    return (unsigned short)(r >> 16);
}

// --- kernel 1: codebook fp32 -> bf16, plus e2[k] = sum(codebook[k]^2) -------
__global__ void k_prep_codebook(const float* __restrict__ cb,
                                unsigned short* __restrict__ cb_bf,
                                float* __restrict__ e2) {
    const int k = blockIdx.x;           // 8192 blocks
    const int c = threadIdx.x;          // 256 threads
    float v = cb[k * CDIM + c];
    cb_bf[k * CDIM + c] = f32_to_bf16_rne(v);
    float s = v * v;
#pragma unroll
    for (int off = 16; off > 0; off >>= 1) s += __shfl_xor(s, off, 32);
    __shared__ float ls[8];
    if ((threadIdx.x & 31) == 0) ls[threadIdx.x >> 5] = s;
    __syncthreads();
    if (threadIdx.x == 0) {
        float t = 0.f;
#pragma unroll
        for (int i = 0; i < 8; ++i) t += ls[i];
        e2[k] = t;
    }
}

// --- kernel 2: z [B,C,H,W] fp32 -> z_bf row-major [N][C] bf16 ---------------
__global__ void k_prep_z(const float* __restrict__ z,
                         unsigned short* __restrict__ z_bf) {
    const int b  = blockIdx.x >> 8;     // 4096 blocks = 16 * 256
    const int c  = blockIdx.x & 255;
    const int hw = threadIdx.x;         // 1024 threads
    float v = z[b * 262144 + c * 1024 + hw];
    z_bf[(b * 1024 + hw) * CDIM + c] = f32_to_bf16_rne(v);
}

// --- kernel 3: WMMA GEMM fused with argmin over score = e2 - 2*z.e ----------
// Block = 256 threads = 8 waves; each wave owns 16 rows (128 rows/block).
// B tiles (16 codebook entries x 256 bf16 = 8KB) staged into LDS with async
// global->LDS loads, triple-buffered, one workgroup barrier per iteration.
__global__ void __launch_bounds__(256)
k_dist_argmin(const unsigned short* __restrict__ z_bf,
              const unsigned short* __restrict__ cb_bf,
              const float* __restrict__ e2,
              int* __restrict__ best_idx,
              float* __restrict__ out_idx_f) {
    __shared__ __align__(16) unsigned short lbuf[3][BUF_US];

    const int tid  = threadIdx.x;
    const int lane = tid & 31;
    const int wave = tid >> 5;
    const int half = lane >> 4;         // K-half per ISA 16-bit operand layout
    const int m    = lane & 15;         // A row within tile / B column (N)
    const int row0 = blockIdx.x * 128 + wave * 16;

    // ---- async staging: thread t copies 32B of the 8KB tile ----------------
    const int sg_n   = tid >> 4;        // entry 0..15 within tile
    const int sg_seg = tid & 15;        // 32B segment within the 512B row
    const unsigned lds_base =
        (unsigned)(size_t)(&lbuf[0][0]) + (unsigned)(sg_n * (ROW_US * 2) + sg_seg * 32);
    const unsigned short* sg_src0 = cb_bf + (size_t)sg_n * CDIM + sg_seg * 16;

    auto stage = [&](int buf, int kk) {
        unsigned dst = lds_base + (unsigned)(buf * BUF_BYTES);
        const unsigned short* src = sg_src0 + (size_t)kk * CDIM;
        asm volatile(
            "global_load_async_to_lds_b128 %0, %1, off\n\t"
            "global_load_async_to_lds_b128 %0, %1, off offset:16"
            :: "v"(dst), "v"(src) : "memory");
    };

    // ---- A operand: 16 rows x 256 channels, resident in 64 VGPRs -----------
    // ISA 16-bit A 16x32 layout: lane m, e=0..7 -> K = half*8+e ;
    //                            e=8..15 -> K = 16 + half*8 + (e-8).
    union AB { v16bf v; uint4 q[2]; };
    AB a[8];
    const uint4* zrow = (const uint4*)(z_bf + (size_t)(row0 + m) * CDIM);
#pragma unroll
    for (int kc = 0; kc < 8; ++kc) {
        a[kc].q[0] = zrow[kc * 4 + half];       // K = kc*32 + half*8 + 0..7
        a[kc].q[1] = zrow[kc * 4 + 2 + half];   // K = kc*32 + 16 + half*8 + 0..7
    }

    float minv[8];
    int   mini[8];
#pragma unroll
    for (int v = 0; v < 8; ++v) { minv[v] = 3.4e38f; mini[v] = 0; }

    // Pipeline prologue: stage tiles 0 and 1.
    stage(0, 0);
    stage(1, 16);

    int p = 0;                          // buffer computed this iteration
    for (int it = 0; it < NIT; ++it) {
        const int kk = it * 16;

        // Own group for buffer p is 2 groups back in a 2-deep pipeline:
        // <=2 outstanding async loads means group(it) has landed.
        if (it + 1 < NIT) {
            asm volatile("s_wait_asynccnt 0x2" ::: "memory");
        } else {
            asm volatile("s_wait_asynccnt 0x0" ::: "memory");
        }
        __syncthreads();                // all waves' tile(it) visible in LDS

        // Stage tile it+2 into the buffer last read at iteration it-1
        // (safe: those reads completed before the barrier above).
        if (it + 2 < NIT) {
            int nb = p + 2; if (nb >= 3) nb -= 3;
            stage(nb, kk + 32);
        }

        // B operand from LDS: lane holds column N = m; e=0..7 -> K = half*8+e,
        // e=8..15 -> K = 16+half*8+(e-8)  -> two ds_load_b128 per chunk.
        const uint4* brow =
            (const uint4*)((const char*)&lbuf[0][0] + p * BUF_BYTES + m * (ROW_US * 2));
        v8f acc = {0.f, 0.f, 0.f, 0.f, 0.f, 0.f, 0.f, 0.f};
#pragma unroll
        for (int kc = 0; kc < 8; ++kc) {
            AB b;
            b.q[0] = brow[kc * 4 + half];
            b.q[1] = brow[kc * 4 + 2 + half];
            acc = __builtin_amdgcn_wmma_f32_16x16x32_bf16(
                false, a[kc].v, false, b.v, (short)0, acc, false, false);
        }

        // D layout: lane L holds N = L&15; VGPR v holds M = 8*(L>>4) + v.
        const float e2n = e2[kk + m];
#pragma unroll
        for (int v = 0; v < 8; ++v) {
            float score = __builtin_fmaf(-2.0f, acc[v], e2n);
            if (score < minv[v]) { minv[v] = score; mini[v] = kk + m; }
        }

        ++p; if (p >= 3) p = 0;
    }

    // Min-reduce across the 16 lanes of each half (xor stays inside the half).
#pragma unroll
    for (int off = 1; off < 16; off <<= 1) {
#pragma unroll
        for (int v = 0; v < 8; ++v) {
            float ov = __shfl_xor(minv[v], off, 32);
            int   oi = __shfl_xor(mini[v], off, 32);
            if (ov < minv[v] || (ov == minv[v] && oi < mini[v])) {
                minv[v] = ov; mini[v] = oi;
            }
        }
    }
    if (m == 0) {
#pragma unroll
        for (int v = 0; v < 8; ++v) {
            int row = row0 + half * 8 + v;
            best_idx[row]  = mini[v];
            out_idx_f[row] = (float)mini[v];
        }
    }
}

// --- kernel 4: scrambled gather (faithful to the mislabeled reshape) + loss -
__global__ void k_gather_loss(const float* __restrict__ z,
                              const float* __restrict__ cb,
                              const int* __restrict__ best_idx,
                              float* __restrict__ out_zq,
                              float* __restrict__ block_partial) {
    const int o  = blockIdx.x * 256 + threadIdx.x;   // linear over [B,H,W,C]
    const int c2 = o & 255;
    const int w2 = (o >> 8) & 31;
    const int h2 = (o >> 13) & 31;
    const int b  = o >> 18;
    // reshape(B,W,C,H).transpose(0,3,1,2): source flat index into z_q[N,C]
    const int i  = ((b * 32 + w2) * 256 + c2) * 32 + h2;
    const int nn = i >> 8;
    const int cc = i & 255;
    const float q = cb[(size_t)best_idx[nn] * CDIM + cc];
    out_zq[o] = q;                                    // z_q_st value == z_q
    const float zv = z[b * 262144 + c2 * 1024 + h2 * 32 + w2];  // zp[b,h2,w2,c2]
    const float d  = zv - q;
    float s = d * d;
#pragma unroll
    for (int off = 16; off > 0; off >>= 1) s += __shfl_xor(s, off, 32);
    __shared__ float ls[8];
    if ((threadIdx.x & 31) == 0) ls[threadIdx.x >> 5] = s;
    __syncthreads();
    if (threadIdx.x == 0) {
        float t = 0.f;
#pragma unroll
        for (int j = 0; j < 8; ++j) t += ls[j];
        block_partial[blockIdx.x] = t;
    }
}

// --- kernel 5: deterministic final loss reduction ---------------------------
__global__ void k_loss_final(const float* __restrict__ block_partial,
                             float* __restrict__ loss_out) {
    float s = 0.f;
    for (int i = threadIdx.x; i < 16384; i += 256) s += block_partial[i];
#pragma unroll
    for (int off = 16; off > 0; off >>= 1) s += __shfl_xor(s, off, 32);
    __shared__ float ls[8];
    if ((threadIdx.x & 31) == 0) ls[threadIdx.x >> 5] = s;
    __syncthreads();
    if (threadIdx.x == 0) {
        float t = 0.f;
        for (int j = 0; j < 8; ++j) t += ls[j];
        // loss = codebook_loss + 0.25*commitment = 1.25 * mean((zp - z_q)^2)
        loss_out[0] = t * (1.25f / 4194304.0f);
    }
}

extern "C" void kernel_launch(void* const* d_in, const int* in_sizes, int n_in,
                              void* d_out, int out_size, void* d_ws, size_t ws_size,
                              hipStream_t stream) {
    const float* z  = (const float*)d_in[0];   // [16,256,32,32]
    const float* cb = (const float*)d_in[1];   // [8192,256]
    float* out = (float*)d_out;

    char* ws = (char*)d_ws;                            // ~12.2 MiB used
    unsigned short* cb_bf = (unsigned short*)(ws);                      // 4 MiB
    unsigned short* z_bf  = (unsigned short*)(ws + 4194304);            // 8 MiB
    float* e2             = (float*)(ws + 12582912);                    // 32 KiB
    int*   best_idx       = (int*)  (ws + 12615680);                    // 64 KiB
    float* partial        = (float*)(ws + 12681216);                    // 64 KiB

    float* out_zq   = out;               // [B,H,W,C] = 4194304 floats
    float* out_loss = out + ZTOTAL;      // scalar
    float* out_idx  = out + ZTOTAL + 1;  // 16384 indices (as float)

    k_prep_codebook<<<K_ENTRIES, 256, 0, stream>>>(cb, cb_bf, e2);
    k_prep_z       <<<4096, 1024, 0, stream>>>(z, z_bf);
    k_dist_argmin  <<<NVEC / 128, 256, 0, stream>>>(z_bf, cb_bf, e2, best_idx, out_idx);
    k_gather_loss  <<<ZTOTAL / 256, 256, 0, stream>>>(z, cb, best_idx, out_zq, partial);
    k_loss_final   <<<1, 256, 0, stream>>>(partial, out_loss);
}
